// ACloss_75436805587392
// MI455X (gfx1250) — compile-verified, matrix operand order checked by reading, not run on hardware
//
#include <hip/hip_runtime.h>
#include <math.h>

#define HEIGHT 256
#define WIDTH  256
#define NUM_L  19
#define SPLIT  4                                   // blocks per heatmap
#define TPB    256                                 // 8 waves of 32
#define HM_ELEMS (HEIGHT*WIDTH)                    // 65536
#define SPLIT_ELEMS (HM_ELEMS/SPLIT)               // 16384
#define QUADS_PER_SPLIT (SPLIT_ELEMS/4)            // 4096 float4s
#define WMMA_ITERS (QUADS_PER_SPLIT/(4*TPB))       // 4: each iter = 4 float4/thread = 16 elems -> 1 WMMA
#define MAXHM  1024

typedef __attribute__((ext_vector_type(16))) _Float16 v16h;
typedef __attribute__((ext_vector_type(8)))  float    v8f;

// ---------------- Stage 1: streaming L2 partial + per-split argmax ----------------
// One block handles 1/4 of one heatmap. Sum of squared diffs is accumulated on the
// matrix pipe: acc = ones(16x32) * sq(32x16) + acc  => every D entry is a column sum,
// layout-independent; total = sum(all D)/16. Argmax runs on VALU concurrently and is
// finished with a deterministic ds_max_u64 on packed (valbits<<32 | ~idx) keys.
__global__ void __launch_bounds__(TPB) acloss_stage1(
    const float* __restrict__ outp, const float* __restrict__ tgtp,
    float* __restrict__ ssq, unsigned long long* __restrict__ okeys,
    unsigned long long* __restrict__ tkeys)
{
    const int blk = blockIdx.x;
    const int hm  = blk / SPLIT;
    const int sp  = blk % SPLIT;
    const int t   = threadIdx.x;

    const float4* ob = (const float4*)(outp + (size_t)hm * HM_ELEMS);
    const float4* tb = (const float4*)(tgtp + (size_t)hm * HM_ELEMS);

    v16h ones;
    #pragma unroll
    for (int k = 0; k < 16; ++k) ones[k] = (_Float16)1.0f;
    v8f acc = {0.f,0.f,0.f,0.f,0.f,0.f,0.f,0.f};

    float    bo = 0.0f;  unsigned boi = 0xFFFFFFFFu;   // key -> 0 (neutral)
    float    bt = 0.0f;  unsigned bti = 0xFFFFFFFFu;

    #pragma unroll
    for (int i = 0; i < WMMA_ITERS; ++i) {
        v16h sq;
        #pragma unroll
        for (int j = 0; j < 4; ++j) {
            const int quad = sp * QUADS_PER_SPLIT + (i*4 + j) * TPB + t;  // coalesced
            const unsigned e0 = (unsigned)quad * 4u;
            float4 o4 = ob[quad];
            float4 t4 = tb[quad];
            float d0 = o4.x - t4.x, d1 = o4.y - t4.y;
            float d2 = o4.z - t4.z, d3 = o4.w - t4.w;
            sq[j*4+0] = (_Float16)(d0*d0);
            sq[j*4+1] = (_Float16)(d1*d1);
            sq[j*4+2] = (_Float16)(d2*d2);
            sq[j*4+3] = (_Float16)(d3*d3);
            // strict '>' + ascending index order => first occurrence within lane
            if (o4.x > bo) { bo = o4.x; boi = e0+0; }
            if (o4.y > bo) { bo = o4.y; boi = e0+1; }
            if (o4.z > bo) { bo = o4.z; boi = e0+2; }
            if (o4.w > bo) { bo = o4.w; boi = e0+3; }
            if (t4.x > bt) { bt = t4.x; bti = e0+0; }
            if (t4.y > bt) { bt = t4.y; bti = e0+1; }
            if (t4.z > bt) { bt = t4.z; bti = e0+2; }
            if (t4.w > bt) { bt = t4.w; bti = e0+3; }
        }
        // EXEC is all-ones here (uniform control flow, full block)
        acc = __builtin_amdgcn_wmma_f32_16x16x32_f16(
            false, ones, false, sq, (short)0, acc, false, false);
    }

    // Per-lane sum of D fragment, then wave32 shuffle reduce.
    float ls = acc[0]+acc[1]+acc[2]+acc[3]+acc[4]+acc[5]+acc[6]+acc[7];
    #pragma unroll
    for (int off = 16; off > 0; off >>= 1) ls += __shfl_down(ls, off, 32);

    __shared__ float wsum[TPB/32];
    __shared__ unsigned long long okey_s, tkey_s;
    if (t == 0) { okey_s = 0ull; tkey_s = 0ull; }
    __syncthreads();

    const int lane = t & 31, wid = t >> 5;
    if (lane == 0) wsum[wid] = ls;

    // Packed argmax keys: value bits are monotone for vals >= 0; ~idx breaks ties
    // toward the smallest (first) index. ds_max_u64 is exact & order-independent.
    unsigned long long ok = ((unsigned long long)__float_as_uint(bo) << 32)
                          | (unsigned long long)(0xFFFFFFFFu - boi);
    unsigned long long tk = ((unsigned long long)__float_as_uint(bt) << 32)
                          | (unsigned long long)(0xFFFFFFFFu - bti);
    atomicMax(&okey_s, ok);
    atomicMax(&tkey_s, tk);
    __syncthreads();

    if (t == 0) {
        float tot = 0.f;
        #pragma unroll
        for (int w = 0; w < TPB/32; ++w) tot += wsum[w];
        ssq[blk]   = tot * (1.0f/16.0f);   // undo 16x row duplication of D
        okeys[blk] = okey_s;
        tkeys[blk] = tkey_s;
    }
}

// ---------------- Stage 2: tiny finalize (single block) ----------------
__device__ __forceinline__ float ang2(float ay, float ax, float by, float bx) {
    float dot = ay*by + ax*bx;
    float den = sqrtf(ay*ay + ax*ax) * sqrtf(by*by + bx*bx);
    if (den == 0.f) return 0.f;                     // matches where(isnan, 0)
    float c = fminf(1.f, fmaxf(-1.f, dot / den));
    return acosf(c);
}

__global__ void __launch_bounds__(256) acloss_stage2(
    const float* __restrict__ ssq,
    const unsigned long long* __restrict__ okeys,
    const unsigned long long* __restrict__ tkeys,
    float* __restrict__ outv, int B, float invN)
{
    const int NH = B * NUM_L;
    __shared__ float ovy[MAXHM], ovx[MAXHM], tvy[MAXHM], tvx[MAXHM];
    __shared__ float red[256];

    // Combine the SPLIT partials per heatmap; decode centered coords.
    float myssq = 0.f;
    for (int h = threadIdx.x; h < NH; h += blockDim.x) {
        float s = 0.f;
        unsigned long long ok = 0ull, tk = 0ull;
        #pragma unroll
        for (int p = 0; p < SPLIT; ++p) {
            s += ssq[h*SPLIT + p];
            unsigned long long a = okeys[h*SPLIT + p]; if (a > ok) ok = a;
            unsigned long long b = tkeys[h*SPLIT + p]; if (b > tk) tk = b;
        }
        myssq += s;
        unsigned oi = 0xFFFFFFFFu - (unsigned)(ok & 0xFFFFFFFFull);
        unsigned ti = 0xFFFFFFFFu - (unsigned)(tk & 0xFFFFFFFFull);
        ovy[h] = (float)(oi / WIDTH) - (float)HEIGHT * 0.5f;
        ovx[h] = (float)(oi % WIDTH) - (float)WIDTH  * 0.5f;
        tvy[h] = (float)(ti / WIDTH) - (float)HEIGHT * 0.5f;
        tvx[h] = (float)(ti % WIDTH) - (float)WIDTH  * 0.5f;
    }
    __syncthreads();

    // Pair loop over B * 19 * 19 entries (centering cancels for distances).
    float angAcc = 0.f, distAcc = 0.f;
    const int P = B * NUM_L * NUM_L;
    for (int p = threadIdx.x; p < P; p += blockDim.x) {
        int b = p / (NUM_L*NUM_L);
        int r = p - b * (NUM_L*NUM_L);
        int l = r / NUM_L;
        int m = r - l * NUM_L;
        int li = b*NUM_L + l, mi = b*NUM_L + m;
        float oa = ang2(ovy[li], ovx[li], ovy[mi], ovx[mi]);
        float ta = ang2(tvy[li], tvx[li], tvy[mi], tvx[mi]);
        angAcc += fabsf(oa - ta);
        float ody = ovy[li]-ovy[mi], odx = ovx[li]-ovx[mi];
        float tdy = tvy[li]-tvy[mi], tdx = tvx[li]-tvx[mi];
        distAcc += fabsf(sqrtf(ody*ody + odx*odx) - sqrtf(tdy*tdy + tdx*tdx));
    }

    // Deterministic tree reductions (fixed order).
    float vals[3] = { myssq, angAcc, distAcc };
    float tot[3];
    for (int k = 0; k < 3; ++k) {
        red[threadIdx.x] = vals[k];
        __syncthreads();
        for (int s = 128; s > 0; s >>= 1) {
            if ((int)threadIdx.x < s) red[threadIdx.x] += red[threadIdx.x + s];
            __syncthreads();
        }
        tot[k] = red[0];
        __syncthreads();
    }

    if (threadIdx.x == 0) {
        float l2         = tot[0] * invN;
        float angle_loss = tot[1] * (1.0f / (float)(NUM_L*NUM_L)); // sum_b of per-image means
        float dist_loss  = tot[2] * (1.0f / (float)(NUM_L*NUM_L));
        float w          = 1.0f + angle_loss + logf(dist_loss + 1e-10f);
        outv[0] = l2 * w;        // loss
        outv[1] = l2;            // l2_loss
        outv[2] = w;             // w_loss
        outv[3] = angle_loss;    // angle_loss
        outv[4] = dist_loss;     // dist_loss
    }
}

extern "C" void kernel_launch(void* const* d_in, const int* in_sizes, int n_in,
                              void* d_out, int out_size, void* d_ws, size_t ws_size,
                              hipStream_t stream) {
    const float* outp = (const float*)d_in[0];
    const float* tgtp = (const float*)d_in[1];
    const int total = in_sizes[0];
    const int B  = total / (NUM_L * HEIGHT * WIDTH);
    const int NB = B * NUM_L * SPLIT;

    float* ssq = (float*)d_ws;
    size_t off = ((size_t)NB * sizeof(float) + 7) & ~(size_t)7;
    unsigned long long* okeys = (unsigned long long*)((char*)d_ws + off);
    unsigned long long* tkeys = okeys + NB;

    acloss_stage1<<<NB, TPB, 0, stream>>>(outp, tgtp, ssq, okeys, tkeys);
    acloss_stage2<<<1, 256, 0, stream>>>(ssq, okeys, tkeys, (float*)d_out,
                                         B, 1.0f / (float)total);
}
